// BackboneDenoisingLayer_5832565588250
// MI455X (gfx1250) — matrix operand order repaired and checked by reading, not compiled, not run on hardware
//
#include <hip/hip_runtime.h>
#include <hip/hip_bf16.h>
#include <math.h>

// ---------------------------------------------------------------------------
// BackboneDenoisingLayer forward for MI455X (gfx1250, wave32, WMMA).
// GEMMs: v_wmma_f32_16x16x32_bf16, B panel staged in LDS via async-to-LDS.
// Edge softmax / segment ops: wave-per-edge f32 VALU + lane-striped atomics.
// ---------------------------------------------------------------------------

typedef __bf16 bf16;
typedef __attribute__((ext_vector_type(16))) __bf16 v16bf;
typedef __attribute__((ext_vector_type(8)))  float  v8f;
typedef __attribute__((ext_vector_type(4)))  unsigned int v4u;
typedef int v4i_gl __attribute__((vector_size(16)));   // matches async-LDS builtin pointee

#define N_NODES_C 16384
#define N_EDGES_C 262144
#define N_SEQ_EDGES_C 32768
#define C_S_C 256
#define C_Z_C 64
#define C_HID_C 16
#define HEADS_C 8
#define IPA_CAT_C 896   // 8*16 + 8*8*4 + 8*64

static inline int cdiv_ll(long long a, long long b) { return (int)((a + b - 1) / b); }

// ------------------------------- helpers -----------------------------------

__device__ inline void atomicMaxF32(float* addr, float val) {
    unsigned int* ua = (unsigned int*)addr;
    unsigned int cur = __float_as_uint(*addr);
    while (__uint_as_float(cur) < val) {
        unsigned int old = atomicCAS(ua, cur, __float_as_uint(val));
        if (old == cur) break;
        cur = old;
    }
}

__global__ void fill_f32(float* __restrict__ p, float v, size_t n) {
    size_t i = (size_t)blockIdx.x * blockDim.x + threadIdx.x;
    if (i < n) p[i] = v;
}

__global__ void pack_bf16(const float* __restrict__ in, bf16* __restrict__ out, size_t n) {
    size_t i = (size_t)blockIdx.x * blockDim.x + threadIdx.x;
    if (i < n) out[i] = (bf16)in[i];
}

// W is (K x N) row-major f32; Wt is (N x K) row-major bf16.
__global__ void transpose_pack(const float* __restrict__ W, bf16* __restrict__ Wt, int K, int N) {
    int i = blockIdx.x * blockDim.x + threadIdx.x;
    if (i >= K * N) return;
    int k = i / N, n = i - k * N;
    Wt[(size_t)n * K + k] = (bf16)W[i];
}

// --------------------------------- GEMM ------------------------------------
// C[M x N] = A[M x K](bf16,row-major) @ Bt[N x K](bf16,row-major)^T + bias[N]
// Block = 8 waves sharing one 32-column B strip staged in LDS (async copy).
// Wave w handles output tile rows [ (blockIdx.x*8+w)*16 , +16 ).
// Requires: M % 128 == 0, N % 32 == 0, K % 32 == 0 (all true for this model).

union FragU { v16bf v; v4u q[2]; };

__device__ inline void wait_async0() {
#if __has_builtin(__builtin_amdgcn_s_wait_asynccnt)
    __builtin_amdgcn_s_wait_asynccnt(0);
#else
    asm volatile("s_wait_asynccnt 0" ::: "memory");
#endif
}

template<int K>
__global__ __launch_bounds__(256) void gemm_bf16(
    const bf16* __restrict__ A, const bf16* __restrict__ Bt,
    const float* __restrict__ bias, void* __restrict__ C,
    int M, int N, int relu, int out_bf16)
{
    __shared__ bf16 panel[32 * K];          // B strip: rows = 32 output cols, K deep

    const int tid = threadIdx.x;
    const int nt  = blockIdx.y;
    const bf16* bsrc = Bt + (size_t)(nt * 32) * K;
    constexpr int CHUNKS = (32 * K) / 8;    // 16-byte chunks; multiple of 256 for all K here

#if __has_builtin(__builtin_amdgcn_global_load_async_to_lds_b128)
    for (int i = tid; i < CHUNKS; i += 256)
        __builtin_amdgcn_global_load_async_to_lds_b128(
            (__attribute__((address_space(1))) v4i_gl*)(bsrc + i * 8),
            (__attribute__((address_space(3))) v4i_gl*)(panel + i * 8), 0, 0);
    wait_async0();
#else
    for (int i = tid; i < CHUNKS; i += 256)
        *(v4u*)(panel + i * 8) = *(const v4u*)(bsrc + i * 8);
#endif
    __syncthreads();

    const int wave = tid >> 5;
    const int lane = tid & 31;
    const int mt   = blockIdx.x * 8 + wave;
    const int lrow = lane & 15;
    const int g    = lane >> 4;

    // A fragment: lane holds row (mt*16+lrow); VGPR0-3: K = g*8..g*8+7, VGPR4-7: +16
    const bf16* aptr = A + (size_t)(mt * 16 + lrow) * K + g * 8;
    // B fragment from LDS: lane holds col; VGPR0-3: K = g*8..g*8+7, VGPR4-7: +16
    const bf16* b0 = panel + lrow * K + g * 8;
    const bf16* b1 = b0 + 16 * K;

    v8f acc0 = {0.f,0.f,0.f,0.f,0.f,0.f,0.f,0.f};
    v8f acc1 = {0.f,0.f,0.f,0.f,0.f,0.f,0.f,0.f};

    for (int k0 = 0; k0 < K; k0 += 32) {
        FragU a, fb0, fb1;
        a.q[0]   = *(const v4u*)(aptr + k0);
        a.q[1]   = *(const v4u*)(aptr + k0 + 16);
        fb0.q[0] = *(const v4u*)(b0 + k0);
        fb0.q[1] = *(const v4u*)(b0 + k0 + 16);
        fb1.q[0] = *(const v4u*)(b1 + k0);
        fb1.q[1] = *(const v4u*)(b1 + k0 + 16);
        acc0 = __builtin_amdgcn_wmma_f32_16x16x32_bf16(false, a.v, false, fb0.v,
                                                       (short)0, acc0, false, false);
        acc1 = __builtin_amdgcn_wmma_f32_16x16x32_bf16(false, a.v, false, fb1.v,
                                                       (short)0, acc1, false, false);
    }

    const int col0 = nt * 32 + lrow;
    const float bi0 = bias ? bias[col0]      : 0.f;
    const float bi1 = bias ? bias[col0 + 16] : 0.f;
#pragma unroll
    for (int r = 0; r < 8; ++r) {
        const int row = mt * 16 + g * 8 + r;   // C layout: VGPR r -> M = r + 8*(lane>>4)
        float c0 = acc0[r] + bi0;
        float c1 = acc1[r] + bi1;
        if (relu) { c0 = fmaxf(c0, 0.f); c1 = fmaxf(c1, 0.f); }
        if (out_bf16) {
            bf16* o = (bf16*)C + (size_t)row * N;
            o[col0] = (bf16)c0; o[col0 + 16] = (bf16)c1;
        } else {
            float* o = (float*)C + (size_t)row * N;
            o[col0] = c0; o[col0 + 16] = c1;
        }
    }
}

static void launch_gemm(const bf16* A, const bf16* Bt, const float* bias, void* C,
                        int M, int N, int K, int relu, int obf, hipStream_t st)
{
    dim3 grid(M / 128, N / 32);
    if (K == 64)
        gemm_bf16<64><<<grid, 256, 0, st>>>(A, Bt, bias, C, M, N, relu, obf);
    else if (K == 256)
        gemm_bf16<256><<<grid, 256, 0, st>>>(A, Bt, bias, C, M, N, relu, obf);
    else
        gemm_bf16<896><<<grid, 256, 0, st>>>(A, Bt, bias, C, M, N, relu, obf);
}

// ------------------------------ LayerNorm ----------------------------------
// One wave per row. out = ((a + resid*mr) - mean)*inv*w + b, then *mask_post.

template<int L>
__global__ void ln_rows(const float* __restrict__ a, const float* __restrict__ resid,
                        const float* __restrict__ mask_resid,
                        const float* __restrict__ w, const float* __restrict__ b,
                        const float* __restrict__ mask_post,
                        float* __restrict__ out, int rows)
{
    const int gtid = blockIdx.x * blockDim.x + threadIdx.x;
    const int row  = gtid >> 5;
    const int lane = threadIdx.x & 31;
    if (row >= rows) return;
    constexpr int PER = L / 32;
    float x[PER];
    const float mr = mask_resid ? mask_resid[row] : 1.f;
    float s = 0.f;
    for (int i = 0; i < PER; ++i) {
        int idx = lane + i * 32;
        float v = a[(size_t)row * L + idx];
        if (resid) v += resid[(size_t)row * L + idx] * mr;
        x[i] = v; s += v;
    }
    for (int m = 16; m >= 1; m >>= 1) s += __shfl_xor(s, m, 32);
    const float mean = s / (float)L;
    float vs = 0.f;
    for (int i = 0; i < PER; ++i) { float d = x[i] - mean; vs += d * d; }
    for (int m = 16; m >= 1; m >>= 1) vs += __shfl_xor(vs, m, 32);
    const float inv = rsqrtf(vs / (float)L + 1e-5f);
    const float pm  = mask_post ? mask_post[row] : 1.f;
    for (int i = 0; i < PER; ++i) {
        int idx = lane + i * 32;
        out[(size_t)row * L + idx] = ((x[i] - mean) * inv * w[idx] + b[idx]) * pm;
    }
}

// ----------------------------- IPA pieces ----------------------------------

// local points -> global: g = R*l + t.  qp_l: (N,96) [a*32+p], kvp_l: (N,288) [a*96+p]
__global__ void pts_global_kernel(const float* __restrict__ qp_l, const float* __restrict__ kvp_l,
                                  const float* __restrict__ rot, const float* __restrict__ trans,
                                  float* __restrict__ qpg, float* __restrict__ kvpg, int N)
{
    int t = blockIdx.x * blockDim.x + threadIdx.x;
    int n = t >> 7, j = t & 127;
    if (n >= N) return;
    const float* R  = rot + (size_t)n * 9;
    const float* tr = trans + (size_t)n * 3;
    float lx, ly, lz; float* outp;
    if (j < 32) {
        const float* s = qp_l + (size_t)n * 96;
        lx = s[j]; ly = s[32 + j]; lz = s[64 + j];
        outp = qpg + (size_t)n * 96 + j * 3;
    } else {
        int p = j - 32;
        const float* s = kvp_l + (size_t)n * 288;
        lx = s[p]; ly = s[96 + p]; lz = s[192 + p];
        outp = kvpg + (size_t)n * 288 + p * 3;
    }
    outp[0] = R[0]*lx + R[1]*ly + R[2]*lz + tr[0];
    outp[1] = R[3]*lx + R[4]*ly + R[5]*lz + tr[1];
    outp[2] = R[6]*lx + R[7]*ly + R[8]*lz + tr[2];
}

__global__ void headw_kernel(const float* __restrict__ hw, float* __restrict__ out) {
    int h = threadIdx.x;
    if (h < HEADS_C) out[h] = log1pf(expf(hw[h])) * 0.13608276348795434f; // sqrt(1/54)
}

// b[e,h] = z[e] . wb[:,h] + bb[h]
__global__ void bproj_kernel(const float* __restrict__ z, const float* __restrict__ wb,
                             const float* __restrict__ bb, float* __restrict__ bedge, int E)
{
    int t = blockIdx.x * blockDim.x + threadIdx.x;
    if (t >= E * HEADS_C) return;
    int e = t >> 3, h = t & 7;
    float acc = bb[h];
    const float* zr = z + (size_t)e * C_Z_C;
    for (int c = 0; c < C_Z_C; ++c) acc += zr[c] * wb[c * HEADS_C + h];
    bedge[t] = acc;
}

__global__ void att1_kernel(const int* __restrict__ ei, int E,
    const float* __restrict__ q, const float* __restrict__ kv,
    const float* __restrict__ qpg, const float* __restrict__ kvpg,
    const float* __restrict__ bedge, const float* __restrict__ headw,
    const float* __restrict__ mask,
    float* __restrict__ att, float* __restrict__ amax)
{
    int t = blockIdx.x * blockDim.x + threadIdx.x;
    if (t >= E * HEADS_C) return;
    int e = t >> 3, h = t & 7;
    int row = ei[e], col = ei[E + e];
    const float* qr = q  + (size_t)row * 128 + h * 16;
    const float* kc = kv + (size_t)col * 256 + h * 32;
    float dq = 0.f;
    for (int c = 0; c < 16; ++c) dq += qr[c] * kc[c];
    float a = dq * 0.14433756729740643f + 0.5773502691896258f * bedge[t];
    const float* qp = qpg  + (size_t)row * 96  + h * 12;  // 4 q-points * 3
    const float* kp = kvpg + (size_t)col * 288 + h * 36;  // first 4 of 12 points * 3
    float d2 = 0.f;
    for (int i = 0; i < 12; ++i) { float d = qp[i] - kp[i]; d2 += d * d; }
    a -= 0.5f * headw[h] * d2;
    a += 100000.f * (mask[col] - 1.f);
    att[t] = a;
    atomicMaxF32(&amax[row * 8 + h], a);
}

// One wave per edge: softmax weights live in lanes 0..7, accumulation targets
// are lane-striped so each atomic instruction covers 32 consecutive addresses.
__global__ void att2_kernel(const int* __restrict__ ei, int E,
    const float* __restrict__ kv, const float* __restrict__ kvpg,
    const float* __restrict__ z,
    const float* __restrict__ att, const float* __restrict__ amax,
    float* __restrict__ denom, float* __restrict__ o,
    float* __restrict__ opt, float* __restrict__ opair)
{
    int gtid = blockIdx.x * blockDim.x + threadIdx.x;
    int e    = gtid >> 5;
    int lane = threadIdx.x & 31;
    if (e >= E) return;                      // wave-uniform
    int row = ei[e], col = ei[E + e];
    int hl = lane & 7;
    float wl = expf(att[(size_t)e * 8 + hl] - amax[row * 8 + hl]);
    if (lane < 8) atomicAdd(&denom[row * 8 + lane], wl);

    const float* kvc = kv + (size_t)col * 256;
    float* orow = o + (size_t)row * 128;
#pragma unroll
    for (int it = 0; it < 4; ++it) {         // o: 128 values
        int j = it * 32 + lane;
        int h = j >> 4, c = j & 15;
        float wh = __shfl(wl, h, 32);
        atomicAdd(&orow[j], wh * kvc[h * 32 + 16 + c]);
    }
    const float* vpc = kvpg + (size_t)col * 288;
    float* po = opt + (size_t)row * 192;
#pragma unroll
    for (int it = 0; it < 6; ++it) {         // o_pt: 192 values (8 heads * 8 pts * 3)
        int j = it * 32 + lane;
        int h = j / 24, i = j - h * 24;
        float wh = __shfl(wl, h, 32);
        atomicAdd(&po[j], wh * vpc[h * 36 + 12 + i]);
    }
    const float* zr = z + (size_t)e * C_Z_C;
    float* pr = opair + (size_t)row * 512;
#pragma unroll
    for (int it = 0; it < 16; ++it) {        // o_pair: 512 values
        int j = it * 32 + lane;
        int h = j >> 6, c = j & 63;
        float wh = __shfl(wl, h, 32);
        atomicAdd(&pr[j], wh * zr[c]);
    }
}

// per (node, head): normalize, inverse-rotate points, build cat row (bf16)
__global__ void nodefin_kernel(const float* __restrict__ denom, const float* __restrict__ o,
    const float* __restrict__ opt, const float* __restrict__ opair,
    const float* __restrict__ rot, const float* __restrict__ trans,
    bf16* __restrict__ cat, int N)
{
    int t = blockIdx.x * blockDim.x + threadIdx.x;
    if (t >= N * HEADS_C) return;
    int n = t >> 3, h = t & 7;
    float inv = 1.f / (denom[t] + 1e-9f);
    bf16* cr = cat + (size_t)n * IPA_CAT_C;
    const float* orow = o + (size_t)n * 128 + h * 16;
    for (int c = 0; c < 16; ++c) cr[h * 16 + c] = (bf16)(orow[c] * inv);
    const float* R  = rot + (size_t)n * 9;
    const float* tr = trans + (size_t)n * 3;
    const float* po = opt + (size_t)n * 192 + h * 24;
    for (int p = 0; p < 8; ++p) {
        float gx = po[p * 3 + 0] * inv - tr[0];
        float gy = po[p * 3 + 1] * inv - tr[1];
        float gz = po[p * 3 + 2] * inv - tr[2];
        float lx = R[0]*gx + R[3]*gy + R[6]*gz;   // R^T
        float ly = R[1]*gx + R[4]*gy + R[7]*gz;
        float lz = R[2]*gx + R[5]*gy + R[8]*gz;
        cr[128 + h * 24 + p * 3 + 0] = (bf16)lx;
        cr[128 + h * 24 + p * 3 + 1] = (bf16)ly;
        cr[128 + h * 24 + p * 3 + 2] = (bf16)lz;
        cr[320 + h * 8 + p] = (bf16)sqrtf(lx*lx + ly*ly + lz*lz + 1e-8f);
    }
    const float* pr = opair + (size_t)n * 512 + h * 64;
    for (int c = 0; c < 64; ++c) cr[384 + h * 64 + c] = (bf16)(pr[c] * inv);
}

// backbone update: one wave per node
__global__ void final_kernel(const float* __restrict__ s, const float* __restrict__ rot,
    const float* __restrict__ trans, const float* __restrict__ nmask,
    const float* __restrict__ bbw, const float* __restrict__ bbb,
    float* __restrict__ out_s, float* __restrict__ out_rot, float* __restrict__ out_tr, int N)
{
    int gtid = blockIdx.x * blockDim.x + threadIdx.x;
    int n = gtid >> 5;
    int lane = threadIdx.x & 31;
    if (n >= N) return;
    const float* sr = s + (size_t)n * 256;
    float nm = nmask[n];
    float p[6] = {0.f,0.f,0.f,0.f,0.f,0.f};
    for (int i = 0; i < 8; ++i) {
        int idx = lane + i * 32;
        float sv = sr[idx];
        out_s[(size_t)n * 256 + idx] = sv;
        float svn = sv * nm;
        for (int j = 0; j < 6; ++j) p[j] += svn * bbw[idx * 6 + j];
    }
    for (int j = 0; j < 6; ++j)
        for (int m = 16; m >= 1; m >>= 1) p[j] += __shfl_xor(p[j], m, 32);
    if (lane == 0) {
        float u[6];
        for (int j = 0; j < 6; ++j) u[j] = (p[j] + bbb[j]) * nm;
        float invn = rsqrtf(1.f + u[0]*u[0] + u[1]*u[1] + u[2]*u[2]);
        float r = invn, qi = u[0]*invn, qj = u[1]*invn, qk = u[2]*invn;
        float Q[9];
        Q[0] = 1.f - 2.f*(qj*qj + qk*qk); Q[1] = 2.f*(qi*qj - qk*r);        Q[2] = 2.f*(qi*qk + qj*r);
        Q[3] = 2.f*(qi*qj + qk*r);        Q[4] = 1.f - 2.f*(qi*qi + qk*qk); Q[5] = 2.f*(qj*qk - qi*r);
        Q[6] = 2.f*(qi*qk - qj*r);        Q[7] = 2.f*(qj*qk + qi*r);        Q[8] = 1.f - 2.f*(qi*qi + qj*qj);
        const float* R = rot + (size_t)n * 9;
        float* Ro = out_rot + (size_t)n * 9;
        for (int a = 0; a < 3; ++a)
            for (int c = 0; c < 3; ++c)
                Ro[a*3 + c] = R[a*3+0]*Q[0*3+c] + R[a*3+1]*Q[1*3+c] + R[a*3+2]*Q[2*3+c];
        const float* tr = trans + (size_t)n * 3;
        float* To = out_tr + (size_t)n * 3;
        for (int a = 0; a < 3; ++a)
            To[a] = R[a*3+0]*u[3] + R[a*3+1]*u[4] + R[a*3+2]*u[5] + tr[a];
    }
}

// ------------------------------ workspace ----------------------------------

struct WS {
    float *z_sp;
    bf16 *bfA, *bfB;
    float *q, *kv, *qp_l, *kvp_l, *qpg, *kvpg;
    float *bedge, *att, *amax, *denom, *o, *opt, *opair, *upd, *s_cur, *headw;
    bf16 *ew1t, *ew2t, *ew3t;
    bf16 *wq_t[2], *wkv_t[2], *wqp_t[2], *wkvp_t[2], *wo_t[2];
    bf16 *tw1t, *tw2t, *tw3t;
};

struct IpaW {
    const bf16 *wq_t, *wkv_t, *wqp_t, *wkvp_t, *wo_t;
    const float *bq, *bkv, *bqp, *bkvp, *wb, *bb, *hw, *bo;
};

static void run_ipa(const float* s_in, const float* z, const int* eidx, int E,
                    const float* rot, const float* trans, const float* mask,
                    const IpaW& W, const WS& w, hipStream_t st)
{
    const int N = N_NODES_C;
    pack_bf16<<<cdiv_ll((long long)N * 256, 256), 256, 0, st>>>(s_in, w.bfA, (size_t)N * 256);
    launch_gemm(w.bfA, W.wq_t,   W.bq,   w.q,     N, 128, 256, 0, 0, st);
    launch_gemm(w.bfA, W.wkv_t,  W.bkv,  w.kv,    N, 256, 256, 0, 0, st);
    launch_gemm(w.bfA, W.wqp_t,  W.bqp,  w.qp_l,  N,  96, 256, 0, 0, st);
    launch_gemm(w.bfA, W.wkvp_t, W.bkvp, w.kvp_l, N, 288, 256, 0, 0, st);
    pts_global_kernel<<<cdiv_ll((long long)N * 128, 256), 256, 0, st>>>(
        w.qp_l, w.kvp_l, rot, trans, w.qpg, w.kvpg, N);
    headw_kernel<<<1, 32, 0, st>>>(W.hw, w.headw);
    fill_f32<<<cdiv_ll((long long)N * 8,   256), 256, 0, st>>>(w.amax,  -3.0e38f, (size_t)N * 8);
    fill_f32<<<cdiv_ll((long long)N * 8,   256), 256, 0, st>>>(w.denom, 0.f, (size_t)N * 8);
    fill_f32<<<cdiv_ll((long long)N * 128, 256), 256, 0, st>>>(w.o,     0.f, (size_t)N * 128);
    fill_f32<<<cdiv_ll((long long)N * 192, 256), 256, 0, st>>>(w.opt,   0.f, (size_t)N * 192);
    fill_f32<<<cdiv_ll((long long)N * 512, 256), 256, 0, st>>>(w.opair, 0.f, (size_t)N * 512);
    bproj_kernel<<<cdiv_ll((long long)E * 8, 256), 256, 0, st>>>(z, W.wb, W.bb, w.bedge, E);
    att1_kernel<<<cdiv_ll((long long)E * 8, 256), 256, 0, st>>>(
        eidx, E, w.q, w.kv, w.qpg, w.kvpg, w.bedge, w.headw, mask, w.att, w.amax);
    att2_kernel<<<cdiv_ll((long long)E * 32, 256), 256, 0, st>>>(
        eidx, E, w.kv, w.kvpg, z, w.att, w.amax, w.denom, w.o, w.opt, w.opair);
    nodefin_kernel<<<cdiv_ll((long long)N * 8, 256), 256, 0, st>>>(
        w.denom, w.o, w.opt, w.opair, rot, trans, w.bfB, N);
    launch_gemm(w.bfB, W.wo_t, W.bo, w.upd, N, 256, IPA_CAT_C, 0, 0, st);
}

// ------------------------------- driver ------------------------------------

extern "C" void kernel_launch(void* const* d_in, const int* in_sizes, int n_in,
                              void* d_out, int out_size, void* d_ws, size_t ws_size,
                              hipStream_t stream)
{
    (void)in_sizes; (void)n_in; (void)out_size; (void)ws_size;
    const int N = N_NODES_C, E = N_EDGES_C, ES = N_SEQ_EDGES_C;

    // Input order: setup_inputs() insertion order; nested params flattened
    // depth-first in Python dict insertion order.
    const float* node       = (const float*)d_in[0];
    const float* rot        = (const float*)d_in[1];
    const float* trans      = (const float*)d_in[2];
    const float* edge_f     = (const float*)d_in[3];
    const float* seq_edge_f = (const float*)d_in[4];
    const float* res_mask   = (const float*)d_in[5];
    const float* noise      = (const float*)d_in[6];
    // edge MLP: w1,b1,w2,b2,w3,b3,lnw,lnb  -> 7..14
    const float* ew1 = (const float*)d_in[7];  const float* eb1 = (const float*)d_in[8];
    const float* ew2 = (const float*)d_in[9];  const float* eb2 = (const float*)d_in[10];
    const float* ew3 = (const float*)d_in[11]; const float* eb3 = (const float*)d_in[12];
    const float* elnw = (const float*)d_in[13]; const float* elnb = (const float*)d_in[14];
    // ipa params: wq,bq,wkv,bkv,wqp,bqp,wkvp,bkvp,wb,bb,hw,wo,bo  (sp: 15.., seq: 28..)
    const int ipa_base[2] = {15, 28};
    const float* ln1w = (const float*)d_in[41];
    const float* ln1b = (const float*)d_in[42];
    // transition MLP 43..50
    const float* tw1 = (const float*)d_in[43]; const float* tb1 = (const float*)d_in[44];
    const float* tw2 = (const float*)d_in[45]; const float* tb2 = (const float*)d_in[46];
    const float* tw3 = (const float*)d_in[47]; const float* tb3 = (const float*)d_in[48];
    const float* tlnw = (const float*)d_in[49]; const float* tlnb = (const float*)d_in[50];
    const float* bbw = (const float*)d_in[51];
    const float* bbb = (const float*)d_in[52];
    const int* eidx  = (const int*)d_in[53];
    const int* seidx = (const int*)d_in[54];

    // ---- workspace carve-up ----
    char* base = (char*)d_ws;
    size_t off = 0;
    auto take = [&](size_t bytes) -> void* {
        off = (off + 255) & ~(size_t)255;
        void* p = base + off; off += bytes; return p;
    };
    WS w;
    w.z_sp  = (float*)take((size_t)E * 64 * 4);
    w.bfA   = (bf16*) take((size_t)E * 64 * 2);   // also holds s_bf / h2 (largest user: E*64)
    w.bfB   = (bf16*) take((size_t)E * 64 * 2);   // also holds cat_bf (N*896 < E*64)
    w.q     = (float*)take((size_t)N * 128 * 4);
    w.kv    = (float*)take((size_t)N * 256 * 4);
    w.qp_l  = (float*)take((size_t)N * 96 * 4);
    w.kvp_l = (float*)take((size_t)N * 288 * 4);
    w.qpg   = (float*)take((size_t)N * 96 * 4);
    w.kvpg  = (float*)take((size_t)N * 288 * 4);
    w.bedge = (float*)take((size_t)E * 8 * 4);
    w.att   = (float*)take((size_t)E * 8 * 4);
    w.amax  = (float*)take((size_t)N * 8 * 4);
    w.denom = (float*)take((size_t)N * 8 * 4);
    w.o     = (float*)take((size_t)N * 128 * 4);
    w.opt   = (float*)take((size_t)N * 192 * 4);
    w.opair = (float*)take((size_t)N * 512 * 4);
    w.upd   = (float*)take((size_t)N * 256 * 4);
    w.s_cur = (float*)take((size_t)N * 256 * 4);
    w.headw = (float*)take(256);
    w.ew1t = (bf16*)take(64 * 64 * 2);
    w.ew2t = (bf16*)take(64 * 64 * 2);
    w.ew3t = (bf16*)take(64 * 64 * 2);
    for (int i = 0; i < 2; ++i) {
        w.wq_t[i]   = (bf16*)take((size_t)256 * 128 * 2);
        w.wkv_t[i]  = (bf16*)take((size_t)256 * 256 * 2);
        w.wqp_t[i]  = (bf16*)take((size_t)256 * 96 * 2);
        w.wkvp_t[i] = (bf16*)take((size_t)256 * 288 * 2);
        w.wo_t[i]   = (bf16*)take((size_t)IPA_CAT_C * 256 * 2);
    }
    w.tw1t = (bf16*)take((size_t)256 * 256 * 2);
    w.tw2t = (bf16*)take((size_t)256 * 256 * 2);
    w.tw3t = (bf16*)take((size_t)256 * 256 * 2);

    // ---- pack weights (transposed bf16) ----
    auto tpack = [&](const float* W_, bf16* Wt, int K, int Nn) {
        transpose_pack<<<cdiv_ll((long long)K * Nn, 256), 256, 0, stream>>>(W_, Wt, K, Nn);
    };
    tpack(ew1, w.ew1t, 64, 64); tpack(ew2, w.ew2t, 64, 64); tpack(ew3, w.ew3t, 64, 64);
    IpaW ipaW[2];
    for (int i = 0; i < 2; ++i) {
        int b = ipa_base[i];
        tpack((const float*)d_in[b + 0], (bf16*)w.wq_t[i],   256, 128);
        tpack((const float*)d_in[b + 2], (bf16*)w.wkv_t[i],  256, 256);
        tpack((const float*)d_in[b + 4], (bf16*)w.wqp_t[i],  256, 96);
        tpack((const float*)d_in[b + 6], (bf16*)w.wkvp_t[i], 256, 288);
        tpack((const float*)d_in[b + 11], (bf16*)w.wo_t[i],  IPA_CAT_C, 256);
        ipaW[i].wq_t = w.wq_t[i];   ipaW[i].bq   = (const float*)d_in[b + 1];
        ipaW[i].wkv_t = w.wkv_t[i]; ipaW[i].bkv  = (const float*)d_in[b + 3];
        ipaW[i].wqp_t = w.wqp_t[i]; ipaW[i].bqp  = (const float*)d_in[b + 5];
        ipaW[i].wkvp_t = w.wkvp_t[i]; ipaW[i].bkvp = (const float*)d_in[b + 7];
        ipaW[i].wb = (const float*)d_in[b + 8];
        ipaW[i].bb = (const float*)d_in[b + 9];
        ipaW[i].hw = (const float*)d_in[b + 10];
        ipaW[i].wo_t = w.wo_t[i];
        ipaW[i].bo = (const float*)d_in[b + 12];
    }
    tpack(tw1, w.tw1t, 256, 256); tpack(tw2, w.tw2t, 256, 256); tpack(tw3, w.tw3t, 256, 256);

    // ---- edge embed (spatial edges only) ----
    pack_bf16<<<cdiv_ll((long long)E * 64, 256), 256, 0, stream>>>(edge_f, w.bfA, (size_t)E * 64);
    launch_gemm(w.bfA, w.ew1t, eb1, w.bfB, E, 64, 64, 1, 1, stream);
    launch_gemm(w.bfB, w.ew2t, eb2, w.bfA, E, 64, 64, 1, 1, stream);
    launch_gemm(w.bfA, w.ew3t, eb3, w.z_sp, E, 64, 64, 0, 0, stream);
    ln_rows<64><<<cdiv_ll((long long)E * 32, 256), 256, 0, stream>>>(
        w.z_sp, nullptr, nullptr, elnw, elnb, nullptr, w.z_sp, E);

    // ---- IPA (spatial) + residual LN ----
    run_ipa(node, w.z_sp, eidx, E, rot, trans, res_mask, ipaW[0], w, stream);
    ln_rows<256><<<cdiv_ll((long long)N * 32, 256), 256, 0, stream>>>(
        node, w.upd, res_mask, ln1w, ln1b, nullptr, w.s_cur, N);

    // ---- IPA (sequence, raw seq edge features as z) + residual LN ----
    run_ipa(w.s_cur, seq_edge_f, seidx, ES, rot, trans, res_mask, ipaW[1], w, stream);
    ln_rows<256><<<cdiv_ll((long long)N * 32, 256), 256, 0, stream>>>(
        w.s_cur, w.upd, res_mask, ln1w, ln1b, nullptr, w.s_cur, N);

    // ---- transition MLP + residual LN + post-mask ----
    pack_bf16<<<cdiv_ll((long long)N * 256, 256), 256, 0, stream>>>(w.s_cur, w.bfA, (size_t)N * 256);
    launch_gemm(w.bfA, w.tw1t, tb1, w.bfB, N, 256, 256, 1, 1, stream);
    launch_gemm(w.bfB, w.tw2t, tb2, w.bfA, N, 256, 256, 1, 1, stream);
    launch_gemm(w.bfA, w.tw3t, tb3, w.upd, N, 256, 256, 0, 0, stream);
    ln_rows<256><<<cdiv_ll((long long)N * 32, 256), 256, 0, stream>>>(
        w.s_cur, w.upd, nullptr, tlnw, tlnb, res_mask, w.s_cur, N);

    // ---- backbone update + outputs (s | new_rot | new_trans, concatenated) ----
    float* out_s   = (float*)d_out;
    float* out_rot = out_s + (size_t)N * 256;
    float* out_tr  = out_rot + (size_t)N * 9;
    final_kernel<<<cdiv_ll((long long)N * 32, 256), 256, 0, stream>>>(
        w.s_cur, rot, trans, noise, bbw, bbb, out_s, out_rot, out_tr, N);
}
